// TransformerBlock_19318762897741
// MI455X (gfx1250) — matrix operand order repaired and checked by reading, not compiled, not run on hardware
//
#include <hip/hip_runtime.h>
#include <hip/hip_bf16.h>

typedef __attribute__((ext_vector_type(16))) _Float16 v16h;
typedef __attribute__((ext_vector_type(8)))  float    v8f;

#define BATCH 4
#define NPTS  4096
#define KNN   36
#define DM    128
#define DP    32

// dynamic LDS partition (bytes)
#define SP_OFF  0                         // float  [48*128] = 24576
#define SV_OFF  24576                     // float  [36*128] = 18432
#define SA_OFF  43008                     // f16    [48*128] = 12288
#define SB_OFF  55296                     // f16    [48*128] = 12288
#define SMEM_BYTES 67584

// ---------- kernel 0: weight f32 [out][in] -> f16 packed in WMMA B-fragment order ----
// Bpack element i: j = i&15 (half within lane), lane = (i>>4)&31, kc = (i>>9)&3,
// nt = i>>11.  Lane's fragment covers col = nt*16 + (lane&15),
// k = kc*32 + (lane>>4)*16 + j  (contiguous k order, matching the A loads).
__global__ void wpack_kernel(const float* __restrict__ W, _Float16* __restrict__ Bp) {
    int i = blockIdx.x * 256 + threadIdx.x;   // over 128*128
    int j    = i & 15;
    int lane = (i >> 4) & 31;
    int kc   = (i >> 9) & 3;
    int nt   = i >> 11;
    int col  = nt * 16 + (lane & 15);
    int k    = kc * 32 + (lane >> 4) * 16 + j;
    Bp[i] = (_Float16)W[col * DM + k];        // W is [out][in] = [col][k]
}

// ---------- kernel 1: brute-force KNN, K smallest with stable index tiebreak ----------
__global__ void __launch_bounds__(64) knn_kernel(const float* __restrict__ xyz,
                                                 int* __restrict__ knn) {
    __shared__ float sd[KNN * 64];
    __shared__ int   si[KNN * 64];
    int t = threadIdx.x;
    int q = blockIdx.x * 64 + t;              // global query id in [0, B*N)
    int b = q >> 12, n = q & (NPTS - 1);
    const float* base = xyz + (size_t)b * NPTS * 3;
    float qx = base[n * 3 + 0], qy = base[n * 3 + 1], qz = base[n * 3 + 2];
    for (int j = 0; j < KNN; ++j) { sd[j * 64 + t] = 3.0e38f; si[j * 64 + t] = 0; }
    for (int m = 0; m < NPTS; ++m) {
        float dx = base[m * 3 + 0] - qx;
        float dy = base[m * 3 + 1] - qy;
        float dz = base[m * 3 + 2] - qz;
        float d  = dx * dx + dy * dy + dz * dz;
        if (d < sd[(KNN - 1) * 64 + t]) {
            int j = KNN - 1;
            while (j > 0 && d < sd[(j - 1) * 64 + t]) {
                sd[j * 64 + t] = sd[(j - 1) * 64 + t];
                si[j * 64 + t] = si[(j - 1) * 64 + t];
                --j;
            }
            sd[j * 64 + t] = d;
            si[j * 64 + t] = m;
        }
    }
    for (int j = 0; j < KNN; ++j) knn[(size_t)q * KNN + j] = si[j * 64 + t];
}

// ---------- kernel 2: x = features @ W1^T + b1 ----------
__global__ void __launch_bounds__(128) feat_kernel(const float* __restrict__ feats,
                                                   const float* __restrict__ W1,
                                                   const float* __restrict__ b1,
                                                   float* __restrict__ x) {
    __shared__ float sf[DP];
    int p = blockIdx.x, o = threadIdx.x;
    if (o < DP) sf[o] = feats[(size_t)p * DP + o];
    __syncthreads();
    float acc = b1[o];
#pragma unroll
    for (int i = 0; i < DP; ++i) acc += sf[i] * W1[o * DP + i];
    x[(size_t)p * DM + o] = acc;
}

// ---------- WMMA GEMM: [48 x 128](f16, LDS) @ [128 x 128](f16 pre-packed, global) ----
// Each wave owns one 16-wide column tile (nt == waveId) and sweeps 3 M-tiles.
// B fragments are pre-packed per (nt, kc, lane): each is ONE aligned v16h load
// (2x global_load_b128, coalesced) -- no strided u16 loads, no half-pack moves.
// A layout (contiguous k, as in Bpack): lane&15 = M row; lane hi-half holds K 16-31.
// D layout (32-bit C/D 16x16): vgpr r -> M = r + 8*(lane>=16), N = lane&15.
template <int RELU_F16>
__device__ __forceinline__ void block_gemm(const _Float16* __restrict__ sA,
                                           const _Float16* __restrict__ Bpack,
                                           const float* __restrict__ bias,
                                           float* __restrict__ outF,
                                           _Float16* __restrict__ outH,
                                           int t) {
    const int lane = t & 31, wave = t >> 5;
    const int l15 = lane & 15, lhi = lane >> 4;
    const int col = wave * 16 + l15;
    const float biasCol = bias[col];

    const v16h* bp = (const v16h*)Bpack;        // [nt][kc][lane] fragments
    const v16h b0 = bp[(wave * 4 + 0) * 32 + lane];
    const v16h b1 = bp[(wave * 4 + 1) * 32 + lane];
    const v16h b2 = bp[(wave * 4 + 2) * 32 + lane];
    const v16h b3 = bp[(wave * 4 + 3) * 32 + lane];

    const int kbA  = lhi * 16;
    const int mOff = lhi * 8;
#pragma unroll
    for (int mt = 0; mt < 3; ++mt) {
        const _Float16* ar = sA + (mt * 16 + l15) * DM;
        v8f  c = {};
        v16h a;
#pragma unroll
        for (int j = 0; j < 8; ++j) { a[j] = ar[kbA + j]; a[8 + j] = ar[kbA + 8 + j]; }
        c = __builtin_amdgcn_wmma_f32_16x16x32_f16(false, a, false, b0, (short)0, c, false, false);
        ar += 32;
#pragma unroll
        for (int j = 0; j < 8; ++j) { a[j] = ar[kbA + j]; a[8 + j] = ar[kbA + 8 + j]; }
        c = __builtin_amdgcn_wmma_f32_16x16x32_f16(false, a, false, b1, (short)0, c, false, false);
        ar += 32;
#pragma unroll
        for (int j = 0; j < 8; ++j) { a[j] = ar[kbA + j]; a[8 + j] = ar[kbA + 8 + j]; }
        c = __builtin_amdgcn_wmma_f32_16x16x32_f16(false, a, false, b2, (short)0, c, false, false);
        ar += 32;
#pragma unroll
        for (int j = 0; j < 8; ++j) { a[j] = ar[kbA + j]; a[8 + j] = ar[kbA + 8 + j]; }
        c = __builtin_amdgcn_wmma_f32_16x16x32_f16(false, a, false, b3, (short)0, c, false, false);

#pragma unroll
        for (int r = 0; r < 8; ++r) {
            int m = mt * 16 + mOff + r;
            float v = c[r] + biasCol;
            if (RELU_F16) outH[m * DM + col] = (_Float16)fmaxf(v, 0.0f);
            else          outF[m * DM + col] = v;
        }
    }
}

// ---------- kernel 3: fused per-query transformer block ----------
__global__ void __launch_bounds__(256) pt_main_kernel(
    const float* __restrict__ xyz, const float* __restrict__ x,
    const int* __restrict__ knn,
    const float* __restrict__ Wd1, const float* __restrict__ bd1,
    const _Float16* __restrict__ Wd2p, const float* __restrict__ bd2,
    const _Float16* __restrict__ Wg1p, const float* __restrict__ bg1,
    const _Float16* __restrict__ Wg2p, const float* __restrict__ bg2,
    const float* __restrict__ W2, const float* __restrict__ b2,
    float* __restrict__ outRes, float* __restrict__ outAttn) {
    extern __shared__ char smem[];
    float*    sP = (float*)(smem + SP_OFF);        // 48x128 pos_enc -> attn_pre -> attn
    float*    sV = (float*)(smem + SV_OFF);        // 36x128 gathered x; later v + pos_enc
    _Float16* sA = (_Float16*)(smem + SA_OFF);     // 48x128 f16 A operand
    _Float16* sB = (_Float16*)(smem + SB_OFF);     // 48x128 f16 gamma hidden

    __shared__ float sXn[DM];
    __shared__ float sRes[DM];
    __shared__ float sDin[KNN * 3];
    __shared__ int   sIdx[KNN];
    __shared__ float sQ[3];

    const int p = blockIdx.x;           // b*N + n
    const int b = p >> 12;
    const int t = threadIdx.x;

    // pull the (shared, hot) f16 weights toward this WGP early
    if (t < 64) {
        __builtin_prefetch(Wd2p + t * 256, 0, 0);
        __builtin_prefetch(Wg1p + t * 256, 0, 0);
        __builtin_prefetch(Wg2p + t * 256, 0, 0);
    }

    if (t < KNN) sIdx[t] = knn[(size_t)p * KNN + t];
    if (t < 3)   sQ[t]   = xyz[(size_t)p * 3 + t];
    if (t < DM)  sXn[t]  = x[(size_t)p * DM + t];
    __syncthreads();

    if (t < KNN) {
        int m = b * NPTS + sIdx[t];
        sDin[t * 3 + 0] = sQ[0] - xyz[(size_t)m * 3 + 0];
        sDin[t * 3 + 1] = sQ[1] - xyz[(size_t)m * 3 + 1];
        sDin[t * 3 + 2] = sQ[2] - xyz[(size_t)m * 3 + 2];
    }
    for (int i = t; i < KNN * DM; i += 256) {   // gather v = k = x[idx]
        int k = i >> 7, f = i & 127;
        sV[i] = x[(size_t)(b * NPTS + sIdx[k]) * DM + f];
    }
    __syncthreads();

    // h1 = relu(d_in @ Wd1^T + bd1) -> sA (f16); pad rows zeroed (keeps pads finite)
    for (int i = t; i < 48 * DM; i += 256) {
        int k = i >> 7, o = i & 127;
        float v = 0.0f;
        if (k < KNN) {
            v = bd1[o] + Wd1[o * 3 + 0] * sDin[k * 3 + 0]
                       + Wd1[o * 3 + 1] * sDin[k * 3 + 1]
                       + Wd1[o * 3 + 2] * sDin[k * 3 + 2];
            v = fmaxf(v, 0.0f);
        }
        sA[i] = (_Float16)v;
    }
    __syncthreads();

    block_gemm<0>(sA, Wd2p, bd2, sP, nullptr, t);   // pos_enc -> sP (48 rows)
    __syncthreads();

    // g_in = x[n] - k + pos_enc -> sA (f16); fold sV := v + pos_enc
    for (int i = t; i < 48 * DM; i += 256) {
        int k = i >> 7, o = i & 127;
        if (k < KNN) {
            float g = sXn[o] - sV[i] + sP[i];
            sA[i] = (_Float16)g;
            sV[i] = sV[i] + sP[i];
        } else {
            sA[i] = (_Float16)0.0f;
        }
    }
    __syncthreads();

    block_gemm<1>(sA, Wg1p, bg1, nullptr, sB, t);   // h2 = relu(...) -> sB (f16)
    __syncthreads();
    block_gemm<0>(sB, Wg2p, bg2, sP, nullptr, t);   // attn_pre -> sP
    __syncthreads();

    // softmax over K per channel, weighted sum with (v + pos_enc)
    if (t < DM) {
        const float scale = 0.08838834764831845f;   // 1/sqrt(128)
        float mx = -3.0e38f;
        for (int k = 0; k < KNN; ++k) mx = fmaxf(mx, sP[k * DM + t]);
        float s = 0.0f;
        for (int k = 0; k < KNN; ++k) {
            float e = __expf((sP[k * DM + t] - mx) * scale);
            sP[k * DM + t] = e; s += e;
        }
        float inv = 1.0f / s;
        float r = 0.0f;
        for (int k = 0; k < KNN; ++k) {
            float a = sP[k * DM + t] * inv;
            outAttn[((size_t)p * KNN + k) * DM + t] = a;
            r += a * sV[k * DM + t];
        }
        sRes[t] = r;
    }
    __syncthreads();

    // res = res @ W2^T + b2 + x
    if (t < DM) {
        float acc = b2[t] + sXn[t];
#pragma unroll 4
        for (int f = 0; f < DM; ++f) acc += sRes[f] * W2[t * DM + f];
        outRes[(size_t)p * DM + t] = acc;
    }
}

extern "C" void kernel_launch(void* const* d_in, const int* in_sizes, int n_in,
                              void* d_out, int out_size, void* d_ws, size_t ws_size,
                              hipStream_t stream) {
    (void)in_sizes; (void)n_in; (void)out_size; (void)ws_size;
    const float* feats = (const float*)d_in[0];
    const float* xyz   = (const float*)d_in[1];
    const float* W1    = (const float*)d_in[2];
    const float* b1    = (const float*)d_in[3];
    const float* W2    = (const float*)d_in[4];
    const float* b2    = (const float*)d_in[5];
    const float* Wd1   = (const float*)d_in[6];
    const float* bd1   = (const float*)d_in[7];
    const float* Wd2   = (const float*)d_in[8];
    const float* bd2   = (const float*)d_in[9];
    const float* Wg1   = (const float*)d_in[10];
    const float* bg1   = (const float*)d_in[11];
    const float* Wg2   = (const float*)d_in[12];
    const float* bg2   = (const float*)d_in[13];

    char* ws = (char*)d_ws;
    size_t off = 0;
    int* knn = (int*)(ws + off);        off += (size_t)BATCH * NPTS * KNN * sizeof(int);
    float* xbuf = (float*)(ws + off);   off += (size_t)BATCH * NPTS * DM * sizeof(float);
    _Float16* Wd2p = (_Float16*)(ws + off); off += (size_t)DM * DM * sizeof(_Float16);
    _Float16* Wg1p = (_Float16*)(ws + off); off += (size_t)DM * DM * sizeof(_Float16);
    _Float16* Wg2p = (_Float16*)(ws + off); off += (size_t)DM * DM * sizeof(_Float16);

    float* outRes  = (float*)d_out;
    float* outAttn = outRes + (size_t)BATCH * NPTS * DM;

    (void)hipFuncSetAttribute((const void*)pt_main_kernel,
                              hipFuncAttributeMaxDynamicSharedMemorySize,
                              SMEM_BYTES);

    wpack_kernel<<<(DM * DM) / 256, 256, 0, stream>>>(Wd2, Wd2p);
    wpack_kernel<<<(DM * DM) / 256, 256, 0, stream>>>(Wg1, Wg1p);
    wpack_kernel<<<(DM * DM) / 256, 256, 0, stream>>>(Wg2, Wg2p);
    knn_kernel<<<(BATCH * NPTS) / 64, 64, 0, stream>>>(xyz, knn);
    feat_kernel<<<BATCH * NPTS, 128, 0, stream>>>(feats, W1, b1, xbuf);
    pt_main_kernel<<<BATCH * NPTS, 256, SMEM_BYTES, stream>>>(xyz, xbuf, knn,
                                                              Wd1, bd1, Wd2p, bd2,
                                                              Wg1p, bg1, Wg2p, bg2,
                                                              W2, b2, outRes, outAttn);
}
